// TGN_53068615910211
// MI455X (gfx1250) — matrix operand order repaired and checked by reading, not compiled, not run on hardware
//
#include <hip/hip_runtime.h>

// ---------------------------------------------------------------------------
// TGN forward for MI455X (gfx1250, wave32).
// All matmuls via v_wmma_f32_16x16x32_f16 (f32 accumulate), elementwise in f32.
// mem/agg tables (25.6MB each) are L2-resident (192MB L2) -> gathers/atomic
// scatters stay on-chip; HBM traffic ~200MB => memory roof ~9us @ 23.3TB/s.
// Edge kernel processes 64 edges/block (4 A-tiles per wave) so each packed
// B-fragment is loaded once and reused by 4 WMMAs -> 4x less weight traffic.
// ---------------------------------------------------------------------------

typedef __attribute__((ext_vector_type(16))) _Float16 v16h;
typedef __attribute__((ext_vector_type(8)))  _Float16 v8h;
typedef __attribute__((ext_vector_type(4)))  _Float16 v4h;
typedef __attribute__((ext_vector_type(8)))  float    v8f;

#define NN      50000
#define EE      600000
#define NODE_IN 64
#define EDGE_IN 32
#define MEMD    128
#define MSGD    128
#define XD      288   // 2*MEM + EDGE_IN
#define ETILES  4     // edge tiles (of 16) per block -> 64 edges/block

static __device__ __forceinline__ v8f wmma16(v16h a, v16h b, v8f c) {
  // D = A(16x32 f16) * B(32x16 f16) + C(16x16 f32)
  return __builtin_amdgcn_wmma_f32_16x16x32_f16(false, a, false, b, (short)0, c,
                                                false, false);
}

// Load a 16x32 f16 A-fragment from an LDS tile (row-major, stride in halves).
// ISA 7.12.2 16-bit A layout: lane L holds row M=L%16; halves 0..7 -> K =
// ks*32 + (L/16)*8 + 0..7 ; halves 8..15 -> +16.  Two 16B LDS loads per lane.
static __device__ __forceinline__ v16h load_a_lds(const _Float16* base,
                                                  int stride_h, int ks, int lane) {
  const int m  = lane & 15;
  const int kb = ks * 32 + ((lane >> 4) << 3);
  const _Float16* p = base + m * stride_h + kb;
  v8h lo = *(const v8h*)(p);
  v8h hi = *(const v8h*)(p + 16);
  v16h a;
#pragma unroll
  for (int i = 0; i < 8; ++i) { a[i] = lo[i]; a[i + 8] = hi[i]; }
  return a;
}

static __device__ __forceinline__ v16h load_b_packed(const _Float16* Wp,
                                                     int frag, int lane) {
  return *(const v16h*)(Wp + (((size_t)frag * 32 + lane) << 4));
}

static __device__ __forceinline__ float sigmoidf_(float x) {
  return 1.0f / (1.0f + __expf(-x));
}

// --------------------------- prep kernels ----------------------------------

__global__ void cvt_f16_kernel(const float* __restrict__ in,
                               _Float16* __restrict__ out, int n) {
  int i = blockIdx.x * blockDim.x + threadIdx.x;
  if (i < n) out[i] = (_Float16)in[i];
}

// Repack row-major W[K, Ncols] (f32) into WMMA B-operand fragment order (f16):
// fragment(ks,nt) = 32 lanes x 16 halves, lane holds col N = nt*16 + lane%16,
// half i holds K = ks*32 + (lane/16)*8 + (i%8) + (i/8)*16.  One contiguous 32B load/lane.
__global__ void pack_b_kernel(const float* __restrict__ W,
                              _Float16* __restrict__ out, int K, int Ncols) {
  int idx = blockIdx.x * blockDim.x + threadIdx.x;
  int total = K * Ncols;
  if (idx >= total) return;
  int NT     = Ncols >> 4;
  int frag   = idx >> 9;          // / 512
  int within = idx & 511;
  int lane   = within >> 4;
  int i      = within & 15;
  int ks     = frag / NT;
  int nt     = frag - ks * NT;
  int k      = ks * 32 + ((lane >> 4) << 3) + (i & 7) + ((i >> 3) << 4);
  int n      = (nt << 4) + (lane & 15);
  out[idx]   = (_Float16)W[(size_t)k * Ncols + n];
}

// --------------------------- edge message kernel ---------------------------
// One block = 64 edges (4 tiles of 16), 8 waves; wave w owns cols [16w,16w+16).
// Per k-step: one B-fragment load feeds 4 WMMAs (one per edge tile).
//   x = [mem_h[src] | mem_h[dst] | f16(edge_z)]  (64 x 288, LDS)
//   h = relu(x @ W1 + b1)   (9 k-steps)
//   msg = h @ W2 + b2       (4 k-steps)
//   atomic scatter msg into agg[dst], counts += 1.
__global__ __launch_bounds__(256) void edge_kernel(
    const _Float16* __restrict__ mem_h, const float* __restrict__ edge_z,
    const int* __restrict__ edge_index,
    const _Float16* __restrict__ W1p, const float* __restrict__ b1,
    const _Float16* __restrict__ W2p, const float* __restrict__ b2,
    float* __restrict__ agg, float* __restrict__ counts) {
  __shared__ alignas(16) _Float16 xs[16 * ETILES][XD + 16];    // ~38.9 KB
  __shared__ alignas(16) _Float16 hs[16 * ETILES][MSGD + 16];  // ~18.4 KB
  __shared__ int dsts[16 * ETILES];

  const int tid  = threadIdx.x;
  const int wave = tid >> 5;
  const int lane = tid & 31;
  const int e0   = blockIdx.x * (16 * ETILES);

  // gather: each wave loads 8 edge rows (128+128 halves vectorized + 32 cvt)
  for (int r = wave; r < 16 * ETILES; r += 8) {
    int e   = e0 + r;
    int src = edge_index[e];
    int dst = edge_index[EE + e];
    if (lane == 0) dsts[r] = dst;
    const v4h* ps = (const v4h*)(mem_h + (size_t)src * MEMD);
    const v4h* pd = (const v4h*)(mem_h + (size_t)dst * MEMD);
    *(v4h*)&xs[r][lane * 4]       = ps[lane];
    *(v4h*)&xs[r][128 + lane * 4] = pd[lane];
    xs[r][256 + lane] = (_Float16)edge_z[(size_t)e * EDGE_IN + lane];
  }
  __syncthreads();

  const int nloc  = lane & 15;
  const int ncol  = wave * 16 + nloc;
  const int mbase = (lane < 16) ? 0 : 8;

  // h tiles: load B once per k-step, reuse across the 4 edge tiles
  v8f acc[ETILES];
#pragma unroll
  for (int t = 0; t < ETILES; ++t) acc[t] = (v8f){};
#pragma unroll
  for (int ks = 0; ks < 9; ++ks) {
    v16h b = load_b_packed(W1p, ks * 8 + wave, lane);
#pragma unroll
    for (int t = 0; t < ETILES; ++t) {
      v16h a = load_a_lds(&xs[t * 16][0], XD + 16, ks, lane);
      acc[t] = wmma16(a, b, acc[t]);
    }
  }
  {
    float bn = b1[ncol];
#pragma unroll
    for (int t = 0; t < ETILES; ++t)
#pragma unroll
      for (int r = 0; r < 8; ++r) {
        float v = acc[t][r] + bn;
        hs[t * 16 + mbase + r][ncol] = (_Float16)(v > 0.f ? v : 0.f);
      }
  }
  __syncthreads();

  // msg tiles
  v8f m[ETILES];
#pragma unroll
  for (int t = 0; t < ETILES; ++t) m[t] = (v8f){};
#pragma unroll
  for (int ks = 0; ks < 4; ++ks) {
    v16h b = load_b_packed(W2p, ks * 8 + wave, lane);
#pragma unroll
    for (int t = 0; t < ETILES; ++t) {
      v16h a = load_a_lds(&hs[t * 16][0], MSGD + 16, ks, lane);
      m[t] = wmma16(a, b, m[t]);
    }
  }
  {
    float bn = b2[ncol];
#pragma unroll
    for (int t = 0; t < ETILES; ++t)
#pragma unroll
      for (int r = 0; r < 8; ++r) {
        int dst = dsts[t * 16 + mbase + r];
        unsafeAtomicAdd(&agg[(size_t)dst * MSGD + ncol], m[t][r] + bn);
      }
  }
  if (tid < 16 * ETILES) unsafeAtomicAdd(&counts[dsts[tid]], 1.0f);
}

// --------------------------- node update kernel ----------------------------
// One block = 16 nodes, 8 waves; wave w owns columns [16w,16w+16) of MEM/EMB.
__global__ __launch_bounds__(256) void node_kernel(
    const float* __restrict__ node_x, const float* __restrict__ mem,
    const _Float16* __restrict__ mem_h,
    const float* __restrict__ agg, const float* __restrict__ counts,
    const _Float16* __restrict__ Wmp,   const float* __restrict__ bm,
    const _Float16* __restrict__ Wxp,   const float* __restrict__ bx,
    const _Float16* __restrict__ Whp,   const float* __restrict__ bh,
    const _Float16* __restrict__ Wencp, const float* __restrict__ benc,
    const _Float16* __restrict__ Wrp,   const float* __restrict__ br,
    const float* __restrict__ Wo, const float* __restrict__ bo,
    float* __restrict__ pred_out, float* __restrict__ newmem_out) {
  __shared__ alignas(16) _Float16 s_agg[16][MEMD + 16];
  __shared__ alignas(16) _Float16 s_mem[16][MEMD + 16];
  __shared__ alignas(16) _Float16 s_am [16][MEMD + 16];
  __shared__ alignas(16) _Float16 s_nm [16][MEMD + 16];
  __shared__ alignas(16) _Float16 s_nx [16][NODE_IN + 16];
  __shared__ float pred_part[8][16];

  const int tid  = threadIdx.x;
  const int wave = tid >> 5;
  const int lane = tid & 31;
  const int n0   = blockIdx.x * 16;

  // ---- cooperative loads ----
  for (int t = tid; t < 16 * MEMD; t += 256) {          // agg mean -> f16
    int row = t >> 7, col = t & 127;
    int node = n0 + row;
    float inv = 1.0f / fmaxf(counts[node], 1.0f);
    s_agg[row][col] = (_Float16)(agg[(size_t)node * MEMD + col] * inv);
  }
  {                                                      // mem_h copy (v8h)
    int row = tid >> 4, col8 = (tid & 15) << 3;
    *(v8h*)&s_mem[row][col8] = *(const v8h*)(mem_h + (size_t)(n0 + row) * MEMD + col8);
  }
  for (int t = tid; t < 16 * NODE_IN; t += 256) {        // node_x -> f16
    int row = t >> 6, col = t & 63;
    s_nx[row][col] = (_Float16)node_x[(size_t)(n0 + row) * NODE_IN + col];
  }
  __syncthreads();

  const int nloc  = lane & 15;
  const int ncol  = wave * 16 + nloc;
  const int mbase = (lane < 16) ? 0 : 8;

  // ---- agg_mem = agg_mean @ Wm + bm ----
  v8f am = {};
#pragma unroll
  for (int ks = 0; ks < 4; ++ks)
    am = wmma16(load_a_lds(&s_agg[0][0], MEMD + 16, ks, lane),
                load_b_packed(Wmp, ks * 8 + wave, lane), am);
  {
    float bn = bm[ncol];
#pragma unroll
    for (int r = 0; r < 8; ++r) s_am[mbase + r][ncol] = (_Float16)(am[r] + bn);
  }
  __syncthreads();

  // ---- GRU gates: gx = agg_mem @ Wx, gh = mem @ Wh  (NT=24; r/z/n = nt w,w+8,w+16)
  v8f gxr = {}, gxz = {}, gxn = {}, ghr = {}, ghz = {}, ghn = {}, ge = {};
#pragma unroll
  for (int ks = 0; ks < 4; ++ks) {
    v16h aam = load_a_lds(&s_am[0][0],  MEMD + 16, ks, lane);
    v16h amm = load_a_lds(&s_mem[0][0], MEMD + 16, ks, lane);
    gxr = wmma16(aam, load_b_packed(Wxp, ks * 24 + wave,      lane), gxr);
    gxz = wmma16(aam, load_b_packed(Wxp, ks * 24 + wave + 8,  lane), gxz);
    gxn = wmma16(aam, load_b_packed(Wxp, ks * 24 + wave + 16, lane), gxn);
    ghr = wmma16(amm, load_b_packed(Whp, ks * 24 + wave,      lane), ghr);
    ghz = wmma16(amm, load_b_packed(Whp, ks * 24 + wave + 8,  lane), ghz);
    ghn = wmma16(amm, load_b_packed(Whp, ks * 24 + wave + 16, lane), ghn);
  }
#pragma unroll
  for (int ks = 0; ks < 2; ++ks)                         // enc (K=64)
    ge = wmma16(load_a_lds(&s_nx[0][0], NODE_IN + 16, ks, lane),
                load_b_packed(Wencp, ks * 8 + wave, lane), ge);

  {
    float bxr = bx[ncol], bxz = bx[128 + ncol], bxn = bx[256 + ncol];
    float bhr = bh[ncol], bhz = bh[128 + ncol], bhn = bh[256 + ncol];
    float ben = benc[ncol];
#pragma unroll
    for (int r = 0; r < 8; ++r) {
      int node = n0 + mbase + r;
      float rr  = sigmoidf_(gxr[r] + bxr + ghr[r] + bhr);
      float zz  = sigmoidf_(gxz[r] + bxz + ghz[r] + bhz);
      float nn  = tanhf(gxn[r] + bxn + rr * (ghn[r] + bhn));
      float mv  = mem[(size_t)node * MEMD + ncol];
      float gru = (1.0f - zz) * nn + zz * mv;
      float ev  = tanhf(ge[r] + ben);
      float nm  = 0.9f * gru + 0.1f * ev;
      newmem_out[(size_t)node * MEMD + ncol] = nm;
      s_nm[mbase + r][ncol] = (_Float16)nm;
    }
  }
  __syncthreads();

  // ---- emb = relu(new_mem @ Wr + br); pred = emb @ Wo + bo ----
  v8f eb = {};
#pragma unroll
  for (int ks = 0; ks < 4; ++ks)
    eb = wmma16(load_a_lds(&s_nm[0][0], MEMD + 16, ks, lane),
                load_b_packed(Wrp, ks * 8 + wave, lane), eb);
  {
    float brn = br[ncol], won = Wo[ncol];
#pragma unroll
    for (int r = 0; r < 8; ++r) {
      float e = fmaxf(eb[r] + brn, 0.0f) * won;
      e += __shfl_xor(e, 1); e += __shfl_xor(e, 2);      // reduce within 16-lane
      e += __shfl_xor(e, 4); e += __shfl_xor(e, 8);      // half-wave (wave32)
      if (nloc == 0) pred_part[wave][mbase + r] = e;
    }
  }
  __syncthreads();
  if (tid < 16) {
    float s = bo[0];
#pragma unroll
    for (int w = 0; w < 8; ++w) s += pred_part[w][tid];
    pred_out[n0 + tid] = s;
  }
}

// ------------------------------ launcher -----------------------------------

extern "C" void kernel_launch(void* const* d_in, const int* in_sizes, int n_in,
                              void* d_out, int out_size, void* d_ws, size_t ws_size,
                              hipStream_t stream) {
  const float* node_x = (const float*)d_in[0];
  const float* edge_z = (const float*)d_in[1];
  const float* mem    = (const float*)d_in[2];
  const int*   eidx   = (const int*)d_in[3];
  const float* W1   = (const float*)d_in[4];
  const float* b1   = (const float*)d_in[5];
  const float* W2   = (const float*)d_in[6];
  const float* b2   = (const float*)d_in[7];
  const float* Wm   = (const float*)d_in[8];
  const float* bm   = (const float*)d_in[9];
  const float* Wenc = (const float*)d_in[10];
  const float* benc = (const float*)d_in[11];
  const float* Wx   = (const float*)d_in[12];
  const float* Wh   = (const float*)d_in[13];
  const float* bx   = (const float*)d_in[14];
  const float* bh   = (const float*)d_in[15];
  const float* Wr   = (const float*)d_in[16];
  const float* br   = (const float*)d_in[17];
  const float* Wo   = (const float*)d_in[18];
  const float* bo   = (const float*)d_in[19];

  char* p = (char*)d_ws;
  auto alloc = [&](size_t bytes) {
    char* r = p;
    p += (bytes + 255) & ~(size_t)255;
    return r;
  };
  float*     agg    = (float*)alloc((size_t)NN * MSGD * 4);
  float*     counts = (float*)alloc((size_t)NN * 4);
  _Float16*  mem_h  = (_Float16*)alloc((size_t)NN * MEMD * 2);
  _Float16*  W1p    = (_Float16*)alloc((size_t)XD * MSGD * 2);
  _Float16*  W2p    = (_Float16*)alloc((size_t)MSGD * MSGD * 2);
  _Float16*  Wmp    = (_Float16*)alloc((size_t)MSGD * MEMD * 2);
  _Float16*  Wxp    = (_Float16*)alloc((size_t)MEMD * 384 * 2);
  _Float16*  Whp    = (_Float16*)alloc((size_t)MEMD * 384 * 2);
  _Float16*  Wencp  = (_Float16*)alloc((size_t)NODE_IN * MEMD * 2);
  _Float16*  Wrp    = (_Float16*)alloc((size_t)MEMD * 128 * 2);
  (void)ws_size; (void)in_sizes; (void)n_in; (void)out_size;

  // fresh accumulators every call (graph-capture safe)
  hipMemsetAsync(agg, 0, (size_t)NN * MSGD * 4, stream);
  hipMemsetAsync(counts, 0, (size_t)NN * 4, stream);

  // prep: f16 mem + weight repack into WMMA B-operand order
  cvt_f16_kernel<<<(NN * MEMD + 255) / 256, 256, 0, stream>>>(mem, mem_h, NN * MEMD);
  pack_b_kernel<<<(XD * MSGD + 255) / 256, 256, 0, stream>>>(W1, W1p, XD, MSGD);
  pack_b_kernel<<<(MSGD * MSGD + 255) / 256, 256, 0, stream>>>(W2, W2p, MSGD, MSGD);
  pack_b_kernel<<<(MSGD * MEMD + 255) / 256, 256, 0, stream>>>(Wm, Wmp, MSGD, MEMD);
  pack_b_kernel<<<(MEMD * 384 + 255) / 256, 256, 0, stream>>>(Wx, Wxp, MEMD, 384);
  pack_b_kernel<<<(MEMD * 384 + 255) / 256, 256, 0, stream>>>(Wh, Whp, MEMD, 384);
  pack_b_kernel<<<(NODE_IN * MEMD + 255) / 256, 256, 0, stream>>>(Wenc, Wencp, NODE_IN, MEMD);
  pack_b_kernel<<<(MEMD * 128 + 255) / 256, 256, 0, stream>>>(Wr, Wrp, MEMD, 128);

  // edge MLP + scatter (9375 blocks of 64 edges)
  edge_kernel<<<EE / (16 * ETILES), 256, 0, stream>>>(mem_h, edge_z, eidx,
                                                      W1p, b1, W2p, b2,
                                                      agg, counts);

  // node update + readout (3125 blocks of 16 nodes)
  float* pred_out   = (float*)d_out;        // [N]
  float* newmem_out = (float*)d_out + NN;   // [N,128]
  node_kernel<<<NN / 16, 256, 0, stream>>>(node_x, mem, mem_h, agg, counts,
                                           Wmp, bm, Wxp, bx, Whp, bh,
                                           Wencp, benc, Wrp, br, Wo, bo,
                                           pred_out, newmem_out);
}
